// VectorQuantizer_64132451664479
// MI455X (gfx1250) — matrix operand order repaired and checked by reading, not compiled, not run on hardware
//
#include <hip/hip_runtime.h>

// ---------------- problem constants ----------------
#define EMBED_D 128
#define NUM_K   2048
#define BATCH   16
#define HWSZ    1024              // H*W = 32*32
#define NROWS   16384             // BATCH*HWSZ
#define ROWS    32                // rows (z vectors) per block (2 WMMA row-tiles)
#define NWAVES  8                 // 256 threads / wave32
#define TILES_PER_WAVE 16         // (NUM_K/16) K-tiles / NWAVES

#define LOG2E   1.4426950408889634f
#define LOG2E2  2.8853900817779268f   // 2*log2(e)
#define LN2     0.6931471805599453f

// ---------------- output layout (floats, reference return order) ---------
#define OFF_LOSS 0ull
#define OFF_ZQ   1ull                               // 16*128*32*32 = 2097152
#define OFF_PERP 2097153ull
#define OFF_OH   2097154ull                         // 16*2048*1024 = 33554432
#define OFF_IDX  35651586ull                        // 16384
#define OFF_IH   35667970ull                        // 32768
#define OFF_SH   35700738ull                        // 32768
#define OUT_TOTAL 35733506ull
#define ZERO_FROM 2097152ull      // 16B-aligned start; [ZERO_FROM, OUT_TOTAL)
                                  // covers onehot+idx+hists (zq/perp rewritten)

typedef __attribute__((ext_vector_type(16))) __bf16 v16bf;
typedef __attribute__((ext_vector_type(8)))  __bf16 v8bf;
typedef __attribute__((ext_vector_type(8)))  float  v8f;

struct Bf32 { uint4 lo, hi; };                      // 32B container for v16bf
static_assert(sizeof(Bf32) == 32, "pad");

__device__ __forceinline__ __bf16 f2bf(float x) { return (__bf16)x; }
__device__ __forceinline__ float exp2fast(float x) {
  return __builtin_amdgcn_exp2f(x);                 // raw v_exp_f32
}

// ---------------- kernel: zero fill (vectorized) ----------------
__global__ void k_zero(float* __restrict__ p, size_t n) {
  size_t i = (size_t)blockIdx.x * blockDim.x + threadIdx.x;
  size_t stride = (size_t)gridDim.x * blockDim.x;
  float4* p4 = (float4*)p;
  size_t n4 = n >> 2;
  for (size_t k = i; k < n4; k += stride) p4[k] = float4{0.f, 0.f, 0.f, 0.f};
  if (i == 0)
    for (size_t k = n4 << 2; k < n; ++k) p[k] = 0.0f;
}

// -------- kernel: per-code squared norms, pre-scaled by log2(e) ----------
__global__ __launch_bounds__(256) void k_cc(const float* __restrict__ cb,
                                            float* __restrict__ cc2) {
  int k = blockIdx.x * 256 + threadIdx.x;
  if (k >= NUM_K) return;
  const float4* p = reinterpret_cast<const float4*>(cb + (size_t)k * EMBED_D);
  float s = 0.f;
#pragma unroll
  for (int i = 0; i < EMBED_D / 4; ++i) {
    float4 f = p[i];
    s += f.x * f.x + f.y * f.y + f.z * f.z + f.w * f.w;
  }
  cc2[k] = s * LOG2E;             // base-2 softmax domain
}

// ---------------- kernel: codebook f32 -> bf16 (one-time, L2 resident) ---
__global__ __launch_bounds__(256) void k_cbf16(const float* __restrict__ cb,
                                               __bf16* __restrict__ cbb) {
  int i = (blockIdx.x * 256 + threadIdx.x) * 8;
  if (i >= NUM_K * EMBED_D) return;
  const float4* p = reinterpret_cast<const float4*>(cb + i);
  float4 f0 = p[0], f1 = p[1];
  v8bf v;
  v[0] = f2bf(f0.x); v[1] = f2bf(f0.y); v[2] = f2bf(f0.z); v[3] = f2bf(f0.w);
  v[4] = f2bf(f1.x); v[5] = f2bf(f1.y); v[6] = f2bf(f1.z); v[7] = f2bf(f1.w);
  *reinterpret_cast<v8bf*>(cbb + i) = v;
}

// ---------------- main fused VQ kernel ----------------
// Per block: 32 rows (one batch b), full K sweep with bf16 WMMA.
// Base-2 logits l2 = log2e*(2*(z.c) - ||c||^2) are bounded (|l2| < ~0.2),
// so exp2(l2) is safe without max-shifting -> one fma + one v_exp per logit.
// acc[0] = sum (zq-z)^2 ; acc[1] = sum_n ln2*(log2 S_n - l2max,n)  (CE loss)
__global__ __launch_bounds__(256) void k_vq_main(
    const float* __restrict__ z, const float* __restrict__ cb,
    const __bf16* __restrict__ cbb, const float* __restrict__ cc2,
    float* __restrict__ out, float* __restrict__ acc) {
  __shared__ float  s_zT[EMBED_D][ROWS];    // 16 KB, async-load dest (d-major)
  __shared__ __bf16 s_zb[ROWS][EMBED_D];    // 8 KB, A-fragment source (r-major)
  __shared__ float  s_ps[NWAVES][ROWS];     // per-wave sum exp2
  __shared__ float  s_pv[NWAVES][ROWS];     // per-wave best logit
  __shared__ int    s_pk[NWAVES][ROWS];     // per-wave best code
  __shared__ float  s_rinv[ROWS];           // 1 / S_row
  __shared__ int    s_rk[ROWS];             // argmin code per row

  const int tid  = threadIdx.x;
  const int wave = tid >> 5;
  const int lane = tid & 31;
  const int kh   = lane >> 4;               // K-half selector (wave32 WMMA layout)
  const int ln   = lane & 15;

  const int n0  = blockIdx.x * ROWS;        // first flat row of this block
  const int b   = n0 >> 10;                 // batch (1024 rows per batch)
  const int hw0 = n0 & 1023;

  // ---- async global->LDS: z tile, transposed so 16B global chunks are
  //      LDS-contiguous. zf[n][d] = z[b, d, hw]; chunk = 4 consecutive hw. ----
  {
    const float* zb = z + (size_t)b * EMBED_D * HWSZ + hw0;
#pragma unroll
    for (int it = 0; it < 4; ++it) {
      int id = tid + 256 * it;              // 1024 chunks of 16B
      int d  = id >> 3;
      int rc = (id & 7) * 4;
      unsigned lds  = (unsigned)(unsigned long long)(const void*)&s_zT[d][rc];
      const void* g = (const void*)(zb + (size_t)d * HWSZ + rc);
      asm volatile("global_load_async_to_lds_b128 %0, %1, off"
                   :: "v"(lds), "v"(g) : "memory");
    }
    asm volatile("s_wait_asynccnt 0x0" ::: "memory");
  }
  __syncthreads();

  // ---- one-time f32 -> bf16 transpose into A-friendly layout ----
  {
    int r  = tid >> 3;
    int dc = (tid & 7) * 16;
#pragma unroll
    for (int i = 0; i < 16; ++i) s_zb[r][dc + i] = f2bf(s_zT[dc + i][r]);
  }
  __syncthreads();

  // ---- build A fragments (two 16-row tiles, contiguous-k 16x32 bf16 lane layout) ----
  // lane (kh,ln): VGPRs0-7 = K 32s+16kh+0..15 (contiguous, matches B)
  v16bf afr[2][4];
#pragma unroll
  for (int t = 0; t < 2; ++t) {
    int m = ln + 16 * t;
#pragma unroll
    for (int s = 0; s < 4; ++s) {
      Bf32 pr;
      pr.lo = *reinterpret_cast<const uint4*>(&s_zb[m][32 * s + 16 * kh]);
      pr.hi = *reinterpret_cast<const uint4*>(&s_zb[m][32 * s + 16 * kh + 8]);
      afr[t][s] = __builtin_bit_cast(v16bf, pr);
    }
  }

  // ---- pass 1: WMMA sweep over K; exp2 softmax + argmax ----
  float run_s[16], bv[16];
  int   bk[16];
#pragma unroll
  for (int j = 0; j < 16; ++j) { run_s[j] = 0.f; bv[j] = -3.0e38f; bk[j] = 0; }

  for (int i = 0; i < TILES_PER_WAVE; ++i) {
    int kbase = (wave + NWAVES * i) << 4;
    int code  = kbase + ln;
    const __bf16* crow = cbb + (size_t)code * EMBED_D + 16 * kh;
    if (i + 1 < TILES_PER_WAVE)
      __builtin_prefetch(cbb + (size_t)(code + NWAVES * 16) * EMBED_D, 0, 3);
    v8f c0 = {}, c1 = {};
#pragma unroll
    for (int s = 0; s < 4; ++s) {           // B: K = 32s+16kh+0..15 contiguous
      v16bf bfr = *reinterpret_cast<const v16bf*>(crow + 32 * s);
      c0 = __builtin_amdgcn_wmma_f32_16x16x32_bf16(false, afr[0][s], false, bfr,
                                                   (short)0, c0, false, false);
      c1 = __builtin_amdgcn_wmma_f32_16x16x32_bf16(false, afr[1][s], false, bfr,
                                                   (short)0, c1, false, false);
    }
    float ccv = cc2[code];
#pragma unroll
    for (int j = 0; j < 8; ++j) {           // rows j+8kh and 16+j+8kh, col ln
      float l0 = __builtin_fmaf(c0[j], LOG2E2, -ccv);
      float l1 = __builtin_fmaf(c1[j], LOG2E2, -ccv);
      run_s[j]     += exp2fast(l0);
      run_s[j + 8] += exp2fast(l1);
      if (l0 > bv[j])     { bv[j] = l0;     bk[j] = code; }
      if (l1 > bv[j + 8]) { bv[j + 8] = l1; bk[j + 8] = code; }
    }
  }

  // ---- cross-lane merge within each 16-lane half (rows shared) ----
#pragma unroll
  for (int j = 0; j < 16; ++j) {
    for (int off = 1; off < 16; off <<= 1) {
      run_s[j] += __shfl_xor(run_s[j], off, 32);
      float ov = __shfl_xor(bv[j], off, 32);
      int   ok = __shfl_xor(bk[j], off, 32);
      if (ov > bv[j] || (ov == bv[j] && ok < bk[j])) { bv[j] = ov; bk[j] = ok; }
    }
  }
  if (ln == 0) {
#pragma unroll
    for (int j = 0; j < 8; ++j) {
      int r0 = j + 8 * kh, r1 = 16 + j + 8 * kh;
      s_ps[wave][r0] = run_s[j];     s_ps[wave][r1] = run_s[j + 8];
      s_pv[wave][r0] = bv[j];        s_pv[wave][r1] = bv[j + 8];
      s_pk[wave][r0] = bk[j];        s_pk[wave][r1] = bk[j + 8];
    }
  }
  __syncthreads();

  // ---- merge 8 wave partials; emit indices / onehot / idx-hist / CE ----
  if (tid < ROWS) {
    int r = tid;
    float S = 0.f, BV = -3.0e38f;
    int BK = 0;
    for (int w = 0; w < NWAVES; ++w) {
      S += s_ps[w][r];
      float v = s_pv[w][r]; int k2 = s_pk[w][r];
      if (v > BV || (v == BV && k2 < BK)) { BV = v; BK = k2; }
    }
    s_rinv[r] = 1.0f / S; s_rk[r] = BK;
    int n = n0 + r, hw = hw0 + r;
    out[OFF_IDX + n] = (float)BK;
    out[OFF_OH + (size_t)b * NUM_K * HWSZ + (size_t)BK * HWSZ + hw] = 1.0f;
    atomicAdd(&out[OFF_IH + (size_t)b * NUM_K + BK], 1.0f);
    atomicAdd(&acc[1], LN2 * (__log2f(S) - BV));  // NLL at argmin
  }
  __syncthreads();

  // ---- zq gather + MSE partial, coalesced: lane = row, wave owns 16 d's ----
  {
    float msum = 0.f;
    int r   = lane;                          // 32 consecutive hw -> coalesced
    int idx = s_rk[r];
    size_t zqb = OFF_ZQ + (size_t)b * EMBED_D * HWSZ + (size_t)(hw0 + r);
    const float4* cb4 = reinterpret_cast<const float4*>(cb);
#pragma unroll
    for (int q = 0; q < 4; ++q) {
      int d0 = wave * 16 + q * 4;
      float4 f = cb4[(size_t)idx * (EMBED_D / 4) + (d0 >> 2)];  // exact f32
      out[zqb + (size_t)(d0 + 0) * HWSZ] = f.x;
      out[zqb + (size_t)(d0 + 1) * HWSZ] = f.y;
      out[zqb + (size_t)(d0 + 2) * HWSZ] = f.z;
      out[zqb + (size_t)(d0 + 3) * HWSZ] = f.w;
      float d0v = f.x - s_zT[d0 + 0][r], d1v = f.y - s_zT[d0 + 1][r];
      float d2v = f.z - s_zT[d0 + 2][r], d3v = f.w - s_zT[d0 + 3][r];
      msum += d0v * d0v + d1v * d1v + d2v * d2v + d3v * d3v;
    }
    for (int off = 1; off < 32; off <<= 1) msum += __shfl_xor(msum, off, 32);
    if (lane == 0) atomicAdd(&acc[0], msum);
  }

  // ---- pass 2: recompute tiles (L2-resident), softmax histogram ----
  for (int i = 0; i < TILES_PER_WAVE; ++i) {
    int kbase = (wave + NWAVES * i) << 4;
    int code  = kbase + ln;
    const __bf16* crow = cbb + (size_t)code * EMBED_D + 16 * kh;
    v8f c0 = {}, c1 = {};
#pragma unroll
    for (int s = 0; s < 4; ++s) {
      v16bf bfr = *reinterpret_cast<const v16bf*>(crow + 32 * s);
      c0 = __builtin_amdgcn_wmma_f32_16x16x32_bf16(false, afr[0][s], false, bfr,
                                                   (short)0, c0, false, false);
      c1 = __builtin_amdgcn_wmma_f32_16x16x32_bf16(false, afr[1][s], false, bfr,
                                                   (short)0, c1, false, false);
    }
    float ccv = cc2[code];
    float psum = 0.f;
#pragma unroll
    for (int j = 0; j < 8; ++j) {
      int r0 = j + 8 * kh;
      psum += exp2fast(__builtin_fmaf(c0[j], LOG2E2, -ccv)) * s_rinv[r0];
      psum += exp2fast(__builtin_fmaf(c1[j], LOG2E2, -ccv)) * s_rinv[r0 + 16];
    }
    psum += __shfl_xor(psum, 16, 32);       // combine the two lane halves
    if (kh == 0)
      atomicAdd(&out[OFF_SH + (size_t)b * NUM_K + code], psum);
  }
}

// ---------------- finalize: loss + perplexity ----------------
__global__ __launch_bounds__(256) void k_vq_final(float* __restrict__ out,
                                                  const float* __restrict__ acc) {
  __shared__ float red[256];
  int tid = threadIdx.x;
  float a = 0.f;
  for (int k = tid; k < NUM_K; k += 256) {
    float cnt = 0.f;
    for (int b = 0; b < BATCH; ++b) cnt += out[OFF_IH + (size_t)b * NUM_K + k];
    float p = cnt * (1.0f / NROWS);
    a += p * __logf(p + 1e-10f);
  }
  red[tid] = a;
  __syncthreads();
  for (int s = 128; s > 0; s >>= 1) {
    if (tid < s) red[tid] += red[tid + s];
    __syncthreads();
  }
  if (tid == 0) {
    out[OFF_PERP] = __expf(-red[0]);
    float mse = acc[0] * (1.0f / ((float)NROWS * EMBED_D));
    // quantization_loss + 0.25*commitment_loss (equal values) + softmax_loss
    out[OFF_LOSS] = 1.25f * mse + acc[1] * (1.0f / NROWS);
  }
}

extern "C" void kernel_launch(void* const* d_in, const int* in_sizes, int n_in,
                              void* d_out, int out_size, void* d_ws, size_t ws_size,
                              hipStream_t stream) {
  (void)in_sizes; (void)n_in; (void)out_size; (void)ws_size;
  const float* z  = (const float*)d_in[0];   // (16,128,32,32) f32
  const float* cb = (const float*)d_in[1];   // (2048,128) f32
  float* out = (float*)d_out;
  float* ws  = (float*)d_ws;
  // ws layout: [0..1] accumulators, [8..2055] log2e*||c||^2, then bf16 codebook
  float*  cc2 = ws + 8;
  __bf16* cbb = (__bf16*)(ws + 8 + NUM_K);   // offset 8224 B (32B aligned)

  // zero only scatter/atomic regions (onehot + hists); rest fully overwritten
  k_zero<<<2048, 256, 0, stream>>>(out + ZERO_FROM, OUT_TOTAL - ZERO_FROM);
  k_zero<<<1, 32, 0, stream>>>(ws, 8);
  k_cc<<<NUM_K / 256, 256, 0, stream>>>(cb, cc2);
  k_cbf16<<<NUM_K * EMBED_D / (8 * 256), 256, 0, stream>>>(cb, cbb);
  k_vq_main<<<NROWS / ROWS, 256, 0, stream>>>(z, cb, cbb, cc2, out, ws);
  k_vq_final<<<1, 256, 0, stream>>>(out, ws);
}
